// T3Attention_54975581388909
// MI455X (gfx1250) — matrix operand order repaired and checked by reading, not compiled, not run on hardware
//
#include <hip/hip_runtime.h>
#include <hip/hip_bf16.h>

typedef __attribute__((ext_vector_type(16))) _Float16     v16h;
typedef __attribute__((ext_vector_type(8)))  _Float16     v8h;
typedef __attribute__((ext_vector_type(8)))  float        v8f;
typedef __attribute__((ext_vector_type(4)))  unsigned int v4u;
typedef __attribute__((ext_vector_type(8)))  int          v8i;
typedef __attribute__((ext_vector_type(4)))  int          v4i;

union V16U { v16h v; v8h h8[2]; };

#define WMMA_F32_F16(a, b, c) \
  __builtin_amdgcn_wmma_f32_16x16x32_f16(false, (a), false, (b), (short)0, (c), false, false)

static constexpr int Bb   = 2;
static constexpr int Ss   = 2048;
static constexpr int Hh   = 2048;
static constexpr int NH   = 32;
static constexpr int NKV  = 8;
static constexpr int Dd   = 64;
static constexpr int QKVN = (NH + 2 * NKV) * Dd;   // 3072
static constexpr float SCALE = 0.125f;             // 1/sqrt(64)

// ---------------------------------------------------------------------------
// Tensor Data Mover: 2D tile load (rows x k-elems, f16) into LDS.
// D# layout per CDNA5 ISA ch. 8 (group0: count/lds/global/type; group1:
// data_size + tensor/tile dims + dim0 stride; group2/3: upper dims unused).
__device__ __forceinline__ void tdm_load_2d_f16(const void* gptr, unsigned lds_addr,
                                                unsigned tile_k, unsigned tile_rows,
                                                unsigned tensor_k, unsigned tensor_rows,
                                                unsigned stride_k) {
  unsigned long long ga = (unsigned long long)(uintptr_t)gptr;
  v4u g0;
  g0.x = 1u;                                             // count=1, user descriptor
  g0.y = lds_addr;                                       // LDS byte address
  g0.z = (unsigned)ga;                                   // global addr [31:0]
  g0.w = (unsigned)((ga >> 32) & 0x01FFFFFFu) | (2u << 30); // addr[56:32] | type=2
  v8i g1;
  g1[0] = (int)(1u << 16);                               // data_size=1 (2 bytes)
  g1[1] = (int)((tensor_k & 0xFFFFu) << 16);             // tensor_dim0[15:0]
  g1[2] = (int)(((tensor_k >> 16) & 0xFFFFu) | ((tensor_rows & 0xFFFFu) << 16));
  g1[3] = (int)(((tensor_rows >> 16) & 0xFFFFu) | ((tile_k & 0xFFFFu) << 16));
  g1[4] = (int)(tile_rows & 0xFFFFu);                    // tile_dim1; tile_dim2=0
  g1[5] = (int)stride_k;                                 // tensor_dim0_stride[31:0]
  g1[6] = 0;
  g1[7] = 0;
  v4i g2 = {1, 1, 0, 0};                                 // tensor_dim2/3 = 1, unused
  v4i g3 = {0, 0, 0, 0};
#if defined(__clang_major__) && __clang_major__ >= 23
  v8i gz = {};
  __builtin_amdgcn_tensor_load_to_lds(g0, g1, g2, g3, gz, 0);
#else
  __builtin_amdgcn_tensor_load_to_lds(g0, g1, g2, g3, 0);
#endif
}

// ---------------------------------------------------------------------------
// fp32 -> f16 elementwise
__global__ void k_convert(const float* __restrict__ in, _Float16* __restrict__ out, int n) {
  int i = blockIdx.x * blockDim.x + threadIdx.x;
  if (i < n) out[i] = (_Float16)in[i];
}

// in [K,N] fp32 row-major -> out [N,K] f16 row-major (so WMMA B-frags are contiguous)
__global__ void k_conv_transpose(const float* __restrict__ in, _Float16* __restrict__ out,
                                 int K, int N) {
  long long i = (long long)blockIdx.x * blockDim.x + threadIdx.x;
  int k = (int)(i % K);
  int n = (int)(i / K);
  out[(size_t)n * K + k] = (_Float16)in[(size_t)k * N + n];
}

// ---------------------------------------------------------------------------
// C[M,N] (f32) = A[M,K] (f16 row-major) x Bt[N,K]^T (f16 row-major-by-N).
// Block = 8 waves -> 128x128 tile; wave = 16x128 (8 accumulators, 8 WMMA/chunk).
// B tile (128 rows x 32 k) staged in LDS by the Tensor Data Mover; each wave
// DMAs its own 16-row slice and waits its own TENSORcnt -> one barrier/chunk.
__global__ __launch_bounds__(256) void k_gemm(const _Float16* __restrict__ A,
                                              const _Float16* __restrict__ Bt,
                                              float* __restrict__ C,
                                              int M, int N, int K) {
  const int lane = threadIdx.x & 31;
  const int wv   = threadIdx.x >> 5;
  const int half = lane >> 4;      // 0: lanes 0-15, 1: lanes 16-31
  const int l15  = lane & 15;
  const int m0 = blockIdx.y * 128 + wv * 16;
  const int n0 = blockIdx.x * 128;

  __shared__ _Float16 Bl[2][128][32];   // 16 KB double buffer

  // A frag (contiguous k order, same as B): lane = row; lanes<16 hold K 0..15, lanes>=16 K 16..31
  const _Float16* ap = A + (size_t)(m0 + l15) * K + half * 16;
  // this wave's TDM slice: 16 rows of the block's B tile
  const _Float16* bslice = Bt + (size_t)(n0 + wv * 16) * K;

  v8f acc[8];
#pragma unroll
  for (int j = 0; j < 8; ++j) acc[j] = (v8f){};

  const int nk = K >> 5;                // 32-wide K chunks
  tdm_load_2d_f16(bslice, (unsigned)(uintptr_t)&Bl[0][wv * 16][0],
                  32u, 16u, (unsigned)K, (unsigned)N, (unsigned)K);

  for (int kc = 0; kc < nk; ++kc) {
    const int k0 = kc << 5;
    __builtin_amdgcn_s_wait_tensorcnt(0);   // own slice of chunk kc is in LDS
    __syncthreads();                        // all 8 slices of buf[kc&1] ready,
                                            // and buf[(kc+1)&1] readers are done
    if (kc + 1 < nk) {
      tdm_load_2d_f16(bslice + (k0 + 32),
                      (unsigned)(uintptr_t)&Bl[(kc + 1) & 1][wv * 16][0],
                      32u, 16u, (unsigned)K, (unsigned)N, (unsigned)K);
    }

    V16U a;
    a.h8[0] = *(const v8h*)(ap + k0);
    a.h8[1] = *(const v8h*)(ap + k0 + 8);
    const int p = kc & 1;
    // B frag (contiguous k order, same as A): lane = col n; lanes<16 hold K 0..15, lanes>=16 K 16..31
    // software-pipeline the LDS fragment reads one step ahead of the WMMAs
    v16h bcur = *(const v16h*)&Bl[p][l15][half * 16];
#pragma unroll
    for (int j = 0; j < 8; ++j) {
      v16h bnxt = bcur;
      if (j < 7) bnxt = *(const v16h*)&Bl[p][(j + 1) * 16 + l15][half * 16];
      acc[j] = WMMA_F32_F16(a.v, bcur, acc[j]);
      bcur = bnxt;
    }
  }

  // C/D layout: VGPR i, lanes<16 -> row m0+i, lanes>=16 -> row m0+i+8; col = n0 + l15 (+16j)
#pragma unroll
  for (int i = 0; i < 8; ++i) {
    float* cr = C + (size_t)(m0 + i + half * 8) * N + n0 + l15;
#pragma unroll
    for (int j = 0; j < 8; ++j) cr[j * 16] = acc[j][i];
  }
}

// ---------------------------------------------------------------------------
// qkv fp32 [B*S, 3072] -> RoPE'd Q f16 [B,NH,S,D], K f16 [B,NKV,S,D], V^T f16 [B,NKV,D,S]
__global__ void k_rope_split(const float* __restrict__ qkv, const int* __restrict__ pos,
                             _Float16* __restrict__ Q, _Float16* __restrict__ Kh,
                             _Float16* __restrict__ Vt) {
  const long long idx = (long long)blockIdx.x * blockDim.x + threadIdx.x;
  const int c = (int)(idx % QKVN);
  const long long r = idx / QKVN;               // row in [0, B*S)
  const int b = (int)(r >> 11);
  const int s = (int)(r & (Ss - 1));
  const float x = qkv[(size_t)r * QKVN + c];
  if (c < (NH + NKV) * Dd) {                    // q or k -> RoPE (NeoX)
    const int base = (c < NH * Dd) ? 0 : NH * Dd;
    const int rel  = c - base;
    const int hh   = rel >> 6;
    const int d    = rel & 63;
    const int p    = pos[s];
    const float j   = (float)(d & 31);
    const float inv = __powf(10000.0f, -j * (1.0f / 32.0f));
    const float ang = (float)p * inv;
    const float cv = __cosf(ang), sv = __sinf(ang);
    const float xp  = qkv[(size_t)r * QKVN + base + hh * 64 + ((d < 32) ? d + 32 : d - 32)];
    const float rot = (d < 32) ? -xp : xp;
    const float y   = x * cv + rot * sv;
    if (c < NH * Dd)
      Q[((size_t)(b * NH + hh) * Ss + s) * Dd + d] = (_Float16)y;
    else
      Kh[((size_t)(b * NKV + hh) * Ss + s) * Dd + d] = (_Float16)y;
  } else {                                      // v -> transpose store
    const int rel = c - (NH + NKV) * Dd;
    const int hh  = rel >> 6;
    const int d   = rel & 63;
    Vt[((size_t)(b * NKV + hh) * Dd + d) * Ss + s] = (_Float16)x;
  }
}

// ---------------------------------------------------------------------------
// Flash attention: one wave per (b, h, 16-query tile), online softmax, causal.
__global__ __launch_bounds__(256) void k_attn(const _Float16* __restrict__ Q,
                                              const _Float16* __restrict__ Kh,
                                              const _Float16* __restrict__ Vt,
                                              _Float16* __restrict__ Ctx) {
  const int lane = threadIdx.x & 31;
  const int wv   = threadIdx.x >> 5;
  const int half = lane >> 4;
  const int l15  = lane & 15;
  const int w  = blockIdx.x * 8 + wv;
  const int qt = w & (Ss / 16 - 1);             // 0..127
  const int h  = (w >> 7) & (NH - 1);
  const int b  = w >> 12;
  const int q0 = qt * 16;
  const int hk = h >> 2;                        // GQA: 4 q-heads per kv-head

  __shared__ _Float16 Pl[8][16][32];            // per-wave P tile (16q x 32k)

  // Q tile as two A-fragments (K-dim 64 -> 2 x 16x32)
  const _Float16* qrow = Q + ((size_t)(b * NH + h) * Ss + q0 + l15) * Dd + half * 16;
  V16U qa0, qa1;
  qa0.h8[0] = *(const v8h*)(qrow);
  qa0.h8[1] = *(const v8h*)(qrow + 8);
  qa1.h8[0] = *(const v8h*)(qrow + 32);
  qa1.h8[1] = *(const v8h*)(qrow + 40);

  const _Float16* kbase = Kh + (size_t)(b * NKV + hk) * Ss * Dd;   // [S,D]
  const _Float16* vbase = Vt + (size_t)(b * NKV + hk) * Dd * Ss;   // [D,S]

  v8f c0 = {}, c1 = {}, c2 = {}, c3 = {};
  float mrow[8], lrow[8];
#pragma unroll
  for (int i = 0; i < 8; ++i) { mrow[i] = -3.0e38f; lrow[i] = 0.0f; }

  const int nch = (q0 + 16 + 31) >> 5;          // 32-key chunks covering keys <= q0+15
  for (int kc = 0; kc < nch; ++kc) {
    const int ka = kc * 32, kb = kc * 32 + 16;
    // scores: B-frag = K^T column n = K row (contiguous [S,D])
    const _Float16* kr0 = kbase + (size_t)(ka + l15) * Dd + half * 16;
    const _Float16* kr1 = kbase + (size_t)(kb + l15) * Dd + half * 16;
    v16h kb00 = *(const v16h*)(kr0);
    v16h kb01 = *(const v16h*)(kr0 + 32);
    v16h kb10 = *(const v16h*)(kr1);
    v16h kb11 = *(const v16h*)(kr1 + 32);
    v8f s0 = {}, s1 = {};
    s0 = WMMA_F32_F16(qa0.v, kb00, s0);
    s0 = WMMA_F32_F16(qa1.v, kb01, s0);
    s1 = WMMA_F32_F16(qa0.v, kb10, s1);
    s1 = WMMA_F32_F16(qa1.v, kb11, s1);

    // online softmax over the 32-key chunk; row i lives across 16 lanes of one VGPR
#pragma unroll
    for (int i = 0; i < 8; ++i) {
      const int qg = q0 + i + half * 8;
      float v0 = (ka + l15 > qg) ? -3.0e38f : s0[i] * SCALE;
      float v1 = (kb + l15 > qg) ? -3.0e38f : s1[i] * SCALE;
      float rm = fmaxf(v0, v1);
#pragma unroll
      for (int off = 1; off < 16; off <<= 1)
        rm = fmaxf(rm, __shfl_xor(rm, off, 16));
      const float mn = fmaxf(mrow[i], rm);
      const float p0 = __expf(v0 - mn);
      const float p1 = __expf(v1 - mn);
      float rs = p0 + p1;
#pragma unroll
      for (int off = 1; off < 16; off <<= 1)
        rs += __shfl_xor(rs, off, 16);
      const float f = __expf(mrow[i] - mn);
      lrow[i] = lrow[i] * f + rs;
      mrow[i] = mn;
      c0[i] *= f; c1[i] *= f; c2[i] *= f; c3[i] *= f;
      Pl[wv][i + half * 8][l15]      = (_Float16)p0;
      Pl[wv][i + half * 8][16 + l15] = (_Float16)p1;
    }
    asm volatile("s_wait_dscnt 0" ::: "memory");   // LDS RAW within wave

    // P as A-fragment (16x32) from LDS
    V16U pa;
    pa.h8[0] = *(const v8h*)&Pl[wv][l15][half * 16];
    pa.h8[1] = *(const v8h*)&Pl[wv][l15][half * 16 + 8];
    // V chunk as B-frags: column n = dim d -> row of V^T (contiguous keys)
    const _Float16* vr = vbase + (size_t)l15 * Ss + ka + half * 16;
    v16h vb0 = *(const v16h*)(vr);
    v16h vb1 = *(const v16h*)(vr + (size_t)16 * Ss);
    v16h vb2 = *(const v16h*)(vr + (size_t)32 * Ss);
    v16h vb3 = *(const v16h*)(vr + (size_t)48 * Ss);
    c0 = WMMA_F32_F16(pa.v, vb0, c0);
    c1 = WMMA_F32_F16(pa.v, vb1, c1);
    c2 = WMMA_F32_F16(pa.v, vb2, c2);
    c3 = WMMA_F32_F16(pa.v, vb3, c3);
  }

  // normalize and store ctx f16 [B*S, NH*D]
#pragma unroll
  for (int i = 0; i < 8; ++i) {
    const float inv = 1.0f / lrow[i];
    _Float16* cr = Ctx + (size_t)(b * Ss + q0 + i + half * 8) * (NH * Dd) + h * Dd + l15;
    cr[0]  = (_Float16)(c0[i] * inv);
    cr[16] = (_Float16)(c1[i] * inv);
    cr[32] = (_Float16)(c2[i] * inv);
    cr[48] = (_Float16)(c3[i] * inv);
  }
}

// ---------------------------------------------------------------------------
extern "C" void kernel_launch(void* const* d_in, const int* in_sizes, int n_in,
                              void* d_out, int out_size, void* d_ws, size_t ws_size,
                              hipStream_t stream) {
  const int*   positions = (const int*)d_in[0];
  const float* hidden    = (const float*)d_in[1];
  const float* Wqkv      = (const float*)d_in[2];
  const float* Wo        = (const float*)d_in[3];
  float*       out       = (float*)d_out;

  char* ws = (char*)d_ws;
  _Float16* Xf16  = (_Float16*)(ws);                     // 16,777,216 B (reused as ctx)
  _Float16* WqkvT = (_Float16*)(ws + 16777216);          // 12,582,912 B
  _Float16* WoT   = (_Float16*)(ws + 29360128);          //  8,388,608 B
  float*    qkvF  = (float*)   (ws + 37748736);          // 50,331,648 B
  _Float16* Qf    = (_Float16*)(ws + 88080384);          // 16,777,216 B
  _Float16* Kf    = (_Float16*)(ws + 104857600);         //  4,194,304 B
  _Float16* Vtf   = (_Float16*)(ws + 109051904);         //  4,194,304 B  (end 113,246,208)
  _Float16* Ctx   = Xf16;                                // X dead after GEMM1

  const int M = Bb * Ss;                                 // 4096

  k_convert<<<(M * Hh) / 256, 256, 0, stream>>>(hidden, Xf16, M * Hh);
  k_conv_transpose<<<(Hh * QKVN) / 256, 256, 0, stream>>>(Wqkv, WqkvT, Hh, QKVN);
  k_conv_transpose<<<(Hh * Hh) / 256, 256, 0, stream>>>(Wo, WoT, Hh, Hh);

  dim3 g1(QKVN / 128, M / 128);
  k_gemm<<<g1, 256, 0, stream>>>(Xf16, WqkvT, qkvF, M, QKVN, Hh);

  k_rope_split<<<(M * QKVN) / 256, 256, 0, stream>>>(qkvF, positions, Qf, Kf, Vtf);

  k_attn<<<(Bb * NH * (Ss / 16)) / 8, 256, 0, stream>>>(Qf, Kf, Vtf, Ctx);

  dim3 g2(Hh / 128, M / 128);
  k_gemm<<<g2, 256, 0, stream>>>(Ctx, WoT, out, M, Hh, Hh);
}
